// ExtractorMLP_37177236914744
// MI455X (gfx1250) — matrix operand order repaired and checked by reading, not compiled, not run on hardware
//
#include <hip/hip_runtime.h>
#include <hip/hip_bf16.h>

// ---------------------------------------------------------------------------
// ExtractorMLP for MI455X (gfx1250): bf16 WMMA GEMMs + fused instance-norm
// stats via L2-resident f32 atomics.  Wave32; v_wmma_f32_16x16x32_bf16.
// Register-blocked: each wave owns a 32x32 output (4 accumulators) so each
// LDS fragment feeds 2 WMMAs -> 8 ds_load_b128 per 4 WMMAs.
// ---------------------------------------------------------------------------

#define H   128     // hidden
#define G   512     // graphs
#define C1  512     // 4H  (layer-1 out channels)
#define K1  256     // 2H  (layer-1 K)
#define EPSI 1e-5f

typedef __attribute__((ext_vector_type(16))) __bf16 v16bf;
typedef __attribute__((ext_vector_type(8)))  float  v8f;
typedef __attribute__((ext_vector_type(4)))  unsigned int u32x4;  // POD 128-bit
typedef __attribute__((ext_vector_type(4)))  float        f32x4;

union Frag { u32x4 q[2]; v16bf v; };   // all POD ext-vectors -> legal union

__device__ __forceinline__ unsigned short f2bf(float f) {
  union { float f; unsigned u; } x; x.f = f;
  unsigned r = (x.u + 0x7FFFu + ((x.u >> 16) & 1u)) >> 16;   // RNE
  return (unsigned short)r;
}
__device__ __forceinline__ float bf2f(unsigned short h) {
  union { unsigned u; float f; } x; x.u = ((unsigned)h) << 16;
  return x.f;
}

#define WMMA_BF16(A, B, C) \
  __builtin_amdgcn_wmma_f32_16x16x32_bf16(false, (A), false, (B), (short)0, (C), false, false)

// ---------------------------------------------------------------------------
__global__ void zero_f32(float* __restrict__ p, int n) {
  int i = blockIdx.x * blockDim.x + threadIdx.x;
  if (i < n) p[i] = 0.0f;
}

// seg[e] = batch[col[e]] ; cnt[g] += 1  (edges per graph, f32 for later math)
__global__ void seg_count(const int* __restrict__ ei, const int* __restrict__ batch,
                          int* __restrict__ seg, float* __restrict__ cnt, int E) {
  int e = blockIdx.x * blockDim.x + threadIdx.x;
  if (e >= E) return;
  int g = batch[ei[e]];
  seg[e] = g;
  atomicAdd(&cnt[g], 1.0f);
}

// mean = sum/clip(n,1); var = sum((x-mean)^2)/clip(n-1,1); rstd = rsqrt(var+eps)
__global__ void finalize_stats(const float* __restrict__ sum, const float* __restrict__ sumsq,
                               const float* __restrict__ cnt, float* __restrict__ mean,
                               float* __restrict__ rstd, int C, int total) {
  int i = blockIdx.x * blockDim.x + threadIdx.x;
  if (i >= total) return;
  int g = i / C;
  float n  = cnt[g];
  float nn = fmaxf(n, 1.0f);
  float m  = sum[i] / nn;
  float ss = fmaxf(sumsq[i] - 2.0f * m * sum[i] + n * m * m, 0.0f);
  float var = ss / fmaxf(n - 1.0f, 1.0f);
  mean[i] = m;
  rstd[i] = rsqrtf(var + EPSI);
}

// ---------------------------------------------------------------------------
// GEMM1: h1[e, n] = concat(emb[col[e]], emb[row[e]]) @ W1 + b1   (K=256)
// block = 256 threads = 8 waves; block tile = 64 edges x 128 channels;
// each wave computes 32x32 (4 independent 16x16 accumulators).
// ---------------------------------------------------------------------------
__global__ void gemm1_wmma(const float* __restrict__ emb, const int* __restrict__ ei,
                           const float* __restrict__ W1, const float* __restrict__ b1,
                           const int* __restrict__ seg, unsigned short* __restrict__ h1,
                           float* __restrict__ sum1, float* __restrict__ sumsq1, int E) {
  extern __shared__ unsigned short smem[];
  const int AP = K1 + 8;                       // 264 halfwords: rows 16B-aligned,
  unsigned short* As = smem;                   //   stride == 4 dwords mod 64 banks
  unsigned short* Bs = smem + 64 * AP;         // Bs[n][k]  (W1 tile, transposed)
  const int tid = threadIdx.x;
  const int m0 = blockIdx.x * 64;
  const int n0 = blockIdx.y * 128;

  // --- A tile: gather 64 edges x 256 cols (f32 -> bf16) -------------------
  {
    int r  = tid >> 2;                 // edge row 0..63
    int c0 = (tid & 3) * 64;           // 64-col chunk (never straddles 128)
    int e  = m0 + r;
    int src = (e < E) ? ((c0 < H) ? ei[e] : ei[E + e]) : 0;
    const float* base = emb + (size_t)src * H + (c0 & (H - 1));
    unsigned short* dst = As + r * AP + c0;
#pragma unroll
    for (int i = 0; i < 64; i += 4) {
      f32x4 f = *(const f32x4*)(base + i);
      dst[i + 0] = f2bf(f.x); dst[i + 1] = f2bf(f.y);
      dst[i + 2] = f2bf(f.z); dst[i + 3] = f2bf(f.w);
    }
  }
  // --- B tile: W1[k, n0+n] -> Bs[n][k] (transposed for contiguous frags) --
  {
    int n  = tid & 127;
    int kc = (tid >> 7) * 128;
    const float* src = W1 + (size_t)kc * C1 + (n0 + n);
    unsigned short* dst = Bs + n * AP + kc;
    for (int k = 0; k < 128; ++k) dst[k] = f2bf(src[(size_t)k * C1]);
  }
  __syncthreads();

  const int w = tid >> 5, lane = tid & 31;
  const int wm = (w >> 2) * 32, wn = (w & 3) * 32;    // 2x4 waves -> 64x128 tile
  const int lh = lane >> 4, l15 = lane & 15;

  float bj0 = b1[n0 + wn + l15];                      // C/D: N = lane%16
  float bj1 = b1[n0 + wn + 16 + l15];
  v8f c00, c01, c10, c11;
#pragma unroll
  for (int r = 0; r < 8; ++r) { c00[r] = bj0; c01[r] = bj1; c10[r] = bj0; c11[r] = bj1; }

  const unsigned short* aR0 = As + (wm + l15) * AP;        // A: M = lane%16
  const unsigned short* aR1 = As + (wm + 16 + l15) * AP;
  const unsigned short* bR0 = Bs + (wn + l15) * AP;        // B: N = lane%16
  const unsigned short* bR1 = Bs + (wn + 16 + l15) * AP;
#pragma unroll
  for (int kk = 0; kk < K1; kk += 32) {
    Frag A0, A1, B0, B1;
    A0.q[0] = *(const u32x4*)(aR0 + kk + lh * 8);        // K {0..7}/{8..15}
    A0.q[1] = *(const u32x4*)(aR0 + kk + 16 + lh * 8);   // K {16..23}/{24..31}
    A1.q[0] = *(const u32x4*)(aR1 + kk + lh * 8);
    A1.q[1] = *(const u32x4*)(aR1 + kk + 16 + lh * 8);
    B0.q[0] = *(const u32x4*)(bR0 + kk + lh * 8);        // K {0..7}/{8..15}
    B0.q[1] = *(const u32x4*)(bR0 + kk + 16 + lh * 8);
    B1.q[0] = *(const u32x4*)(bR1 + kk + lh * 8);
    B1.q[1] = *(const u32x4*)(bR1 + kk + 16 + lh * 8);
    c00 = WMMA_BF16(A0.v, B0.v, c00);
    c01 = WMMA_BF16(A0.v, B1.v, c01);
    c10 = WMMA_BF16(A1.v, B0.v, c10);
    c11 = WMMA_BF16(A1.v, B1.v, c11);
  }

  // --- writeback + per-(graph,channel) stats atomics ----------------------
  v8f* cs[4] = { &c00, &c01, &c10, &c11 };
#pragma unroll
  for (int i = 0; i < 2; ++i) {
#pragma unroll
    for (int j = 0; j < 2; ++j) {
      v8f& c = *cs[i * 2 + j];
      int ch = n0 + wn + j * 16 + l15;
#pragma unroll
      for (int r = 0; r < 8; ++r) {
        int e = m0 + wm + i * 16 + lh * 8 + r;           // C/D: M = r + 8*(lane/16)
        if (e < E) {
          float v = c[r];
          h1[(size_t)e * C1 + ch] = f2bf(v);
          int g = seg[e];
          atomicAdd(&sum1[g * C1 + ch], v);
          atomicAdd(&sumsq1[g * C1 + ch], v * v);
        }
      }
    }
  }
}

// ---------------------------------------------------------------------------
// GEMM2: h2 = relu(IN(h1)) @ W2 + b2   (K=512, normalization fused into the
// LDS fill of the A tile). Same 32x32-per-wave register blocking.
// ---------------------------------------------------------------------------
__global__ void gemm2_wmma(const unsigned short* __restrict__ h1,
                           const float* __restrict__ W2, const float* __restrict__ b2,
                           const int* __restrict__ seg,
                           const float* __restrict__ mean1, const float* __restrict__ rstd1,
                           unsigned short* __restrict__ h2,
                           float* __restrict__ sum2, float* __restrict__ sumsq2, int E) {
  extern __shared__ unsigned short smem[];
  const int K = C1, AP = K + 8;                // 520 halfwords
  unsigned short* As = smem;
  unsigned short* Bs = smem + 64 * AP;
  const int tid = threadIdx.x;
  const int m0 = blockIdx.x * 64;
  const int n0 = blockIdx.y * 128;

  // --- A tile: normalize + relu h1 rows (bf16 in, bf16 out) ---------------
  {
    int r  = tid >> 2;
    int c0 = (tid & 3) * 128;
    int e  = (m0 + r < E) ? (m0 + r) : (E - 1);
    int g  = seg[e];
    const unsigned short* hp = h1 + (size_t)e * K + c0;
    const float* mp = mean1 + g * K + c0;
    const float* rp = rstd1 + g * K + c0;
    unsigned short* dst = As + r * AP + c0;
    for (int i = 0; i < 128; ++i) {
      float x = (bf2f(hp[i]) - mp[i]) * rp[i];
      dst[i] = f2bf(fmaxf(x, 0.0f));
    }
  }
  // --- B tile: W2[k, n0+n] -> Bs[n][k] ------------------------------------
  {
    int n  = tid & 127;
    int kc = (tid >> 7) * 256;
    const float* src = W2 + (size_t)kc * H + (n0 + n);
    unsigned short* dst = Bs + n * AP + kc;
    for (int k = 0; k < 256; ++k) dst[k] = f2bf(src[(size_t)k * H]);
  }
  __syncthreads();

  const int w = tid >> 5, lane = tid & 31;
  const int wm = (w >> 2) * 32, wn = (w & 3) * 32;
  const int lh = lane >> 4, l15 = lane & 15;

  float bj0 = b2[n0 + wn + l15];
  float bj1 = b2[n0 + wn + 16 + l15];
  v8f c00, c01, c10, c11;
#pragma unroll
  for (int r = 0; r < 8; ++r) { c00[r] = bj0; c01[r] = bj1; c10[r] = bj0; c11[r] = bj1; }

  const unsigned short* aR0 = As + (wm + l15) * AP;
  const unsigned short* aR1 = As + (wm + 16 + l15) * AP;
  const unsigned short* bR0 = Bs + (wn + l15) * AP;
  const unsigned short* bR1 = Bs + (wn + 16 + l15) * AP;
#pragma unroll
  for (int kk = 0; kk < K; kk += 32) {
    Frag A0, A1, B0, B1;
    A0.q[0] = *(const u32x4*)(aR0 + kk + lh * 8);
    A0.q[1] = *(const u32x4*)(aR0 + kk + 16 + lh * 8);
    A1.q[0] = *(const u32x4*)(aR1 + kk + lh * 8);
    A1.q[1] = *(const u32x4*)(aR1 + kk + 16 + lh * 8);
    B0.q[0] = *(const u32x4*)(bR0 + kk + lh * 8);
    B0.q[1] = *(const u32x4*)(bR0 + kk + 16 + lh * 8);
    B1.q[0] = *(const u32x4*)(bR1 + kk + lh * 8);
    B1.q[1] = *(const u32x4*)(bR1 + kk + 16 + lh * 8);
    c00 = WMMA_BF16(A0.v, B0.v, c00);
    c01 = WMMA_BF16(A0.v, B1.v, c01);
    c10 = WMMA_BF16(A1.v, B0.v, c10);
    c11 = WMMA_BF16(A1.v, B1.v, c11);
  }

  v8f* cs[4] = { &c00, &c01, &c10, &c11 };
#pragma unroll
  for (int i = 0; i < 2; ++i) {
#pragma unroll
    for (int j = 0; j < 2; ++j) {
      v8f& c = *cs[i * 2 + j];
      int ch = n0 + wn + j * 16 + l15;
#pragma unroll
      for (int r = 0; r < 8; ++r) {
        int e = m0 + wm + i * 16 + lh * 8 + r;
        if (e < E) {
          float v = c[r];
          h2[(size_t)e * H + ch] = f2bf(v);
          int g = seg[e];
          atomicAdd(&sum2[g * H + ch], v);
          atomicAdd(&sumsq2[g * H + ch], v * v);
        }
      }
    }
  }
}

// ---------------------------------------------------------------------------
// out[e] = relu(IN(h2))[e,:] . W3 + b3     (tiny GEMV; memory bound)
// ---------------------------------------------------------------------------
__global__ void final_mlp(const unsigned short* __restrict__ h2,
                          const float* __restrict__ W3, const float* __restrict__ b3,
                          const int* __restrict__ seg,
                          const float* __restrict__ mean2, const float* __restrict__ rstd2,
                          float* __restrict__ out, int E) {
  int e = blockIdx.x * blockDim.x + threadIdx.x;
  if (e >= E) return;
  int g = seg[e];
  const unsigned short* hp = h2 + (size_t)e * H;
  const float* mp = mean2 + g * H;
  const float* rp = rstd2 + g * H;
  float acc = 0.0f;
#pragma unroll 8
  for (int c = 0; c < H; ++c) {
    float x = (bf2f(hp[c]) - mp[c]) * rp[c];
    acc += fmaxf(x, 0.0f) * W3[c];
  }
  out[e] = acc + b3[0];
}

// ---------------------------------------------------------------------------
extern "C" void kernel_launch(void* const* d_in, const int* in_sizes, int n_in,
                              void* d_out, int out_size, void* d_ws, size_t ws_size,
                              hipStream_t stream) {
  (void)n_in; (void)out_size; (void)ws_size;
  const float* emb   = (const float*)d_in[0];
  const int*   ei    = (const int*)  d_in[1];   // [2E] col then row
  const int*   batch = (const int*)  d_in[2];   // [N]
  const float* W1 = (const float*)d_in[3];
  const float* b1 = (const float*)d_in[4];
  const float* W2 = (const float*)d_in[5];
  const float* b2 = (const float*)d_in[6];
  const float* W3 = (const float*)d_in[7];
  const float* b3 = (const float*)d_in[8];
  const int E = in_sizes[1] / 2;
  float* outp = (float*)d_out;

  // Workspace carve (every region is a multiple of 256B at these sizes)
  char* ws = (char*)d_ws;
  auto carve = [&](size_t bytes) {
    char* p = ws; ws += (bytes + 255) & ~(size_t)255; return p;
  };
  int*   seg    = (int*)  carve((size_t)E * 4);
  float* cnt    = (float*)carve((size_t)G * 4);
  float* sum1   = (float*)carve((size_t)G * C1 * 4);
  float* sumsq1 = (float*)carve((size_t)G * C1 * 4);
  float* mean1  = (float*)carve((size_t)G * C1 * 4);
  float* rstd1  = (float*)carve((size_t)G * C1 * 4);
  float* sum2   = (float*)carve((size_t)G * H * 4);
  float* sumsq2 = (float*)carve((size_t)G * H * 4);
  float* mean2  = (float*)carve((size_t)G * H * 4);
  float* rstd2  = (float*)carve((size_t)G * H * 4);
  unsigned short* h1 = (unsigned short*)carve((size_t)E * C1 * 2);
  unsigned short* h2 = (unsigned short*)carve((size_t)E * H * 2);

  size_t sm1 = (size_t)(64 + 128) * (K1 + 8) * 2;    // ~101 KB LDS
  size_t sm2 = (size_t)(64 + 128) * (C1 + 8) * 2;    // ~200 KB LDS (of 320 KB/WGP)
  // Host-side, capture-safe: allow >default dynamic LDS for the GEMM kernels.
  (void)hipFuncSetAttribute((const void*)gemm1_wmma,
                            hipFuncAttributeMaxDynamicSharedMemorySize, (int)sm1);
  (void)hipFuncSetAttribute((const void*)gemm2_wmma,
                            hipFuncAttributeMaxDynamicSharedMemorySize, (int)sm2);

  // cnt..rstd2 are contiguous; zero them all each call (graph-replay safe)
  const int ZT = G + 4 * G * C1 + 4 * G * H;
  zero_f32<<<(ZT + 255) / 256, 256, 0, stream>>>(cnt, ZT);

  seg_count<<<(E + 255) / 256, 256, 0, stream>>>(ei, batch, seg, cnt, E);

  dim3 g1((E + 63) / 64, C1 / 128);
  gemm1_wmma<<<g1, 256, sm1, stream>>>(emb, ei, W1, b1, seg, h1, sum1, sumsq1, E);

  finalize_stats<<<(G * C1 + 255) / 256, 256, 0, stream>>>(sum1, sumsq1, cnt,
                                                           mean1, rstd1, C1, G * C1);

  dim3 g2((E + 63) / 64, H / 128);
  gemm2_wmma<<<g2, 256, sm2, stream>>>(h1, W2, b2, seg, mean1, rstd1,
                                       h2, sum2, sumsq2, E);

  finalize_stats<<<(G * H + 255) / 256, 256, 0, stream>>>(sum2, sumsq2, cnt,
                                                          mean2, rstd2, H, G * H);

  final_mlp<<<(E + 255) / 256, 256, 0, stream>>>(h2, W3, b3, seg, mean2, rstd2, outp, E);
}
